// MemoryEfficientCrossAttention_14345190769331
// MI455X (gfx1250) — matrix operand order repaired and checked by reading, not compiled, not run on hardware
//
#include <hip/hip_runtime.h>

typedef __bf16 bf16;
typedef __attribute__((ext_vector_type(16))) __bf16 v16bf;
typedef __attribute__((ext_vector_type(8)))  __bf16 v8bf;
typedef __attribute__((ext_vector_type(8)))  float  v8f;

#define BATCH    2
#define SEQ_N    4096
#define SEQ_M    4096
#define DMODEL   1024
#define HEADS    16
#define DHEAD    64
#define ROWS     (BATCH * SEQ_N)   // 8192 (== BATCH*SEQ_M too)

// ---------------------------------------------------------------------------
// Fragment helpers (wave32; A and B both use k = 16*half + i)
// ---------------------------------------------------------------------------

// 16 contiguous bf16 -> v16bf (B-fragment piece: lane=col, elems = K=16*half+i)
__device__ inline v16bf ldfrag16(const bf16* p) {
    const v8bf* q = (const v8bf*)p;
    v8bf lo = q[0], hi = q[1];
    v16bf r;
#pragma unroll
    for (int i = 0; i < 8; ++i) { r[i] = lo[i]; r[i + 8] = hi[i]; }
    return r;
}

// A-fragment for a 16x32 tile: rowp points at (row, K-base).
// elems 0..15 -> K = 16*half + i
__device__ inline v16bf ldfragA(const bf16* rowp, int half) {
    const v8bf* q0 = (const v8bf*)(rowp + 16 * half);
    const v8bf* q1 = (const v8bf*)(rowp + 16 * half + 8);
    v8bf lo = q0[0], hi = q1[0];
    v16bf r;
#pragma unroll
    for (int i = 0; i < 8; ++i) { r[i] = lo[i]; r[i + 8] = hi[i]; }
    return r;
}

__device__ inline v8f wmma_bf16(v16bf a, v16bf b, v8f c) {
    return __builtin_amdgcn_wmma_f32_16x16x32_bf16(
        false, a, false, b, (short)0, c, false, false);
}

// width-16 max reduction: stays inside each half-wave (rows 0-7 vs 8-15)
__device__ inline float hmax16(float v) {
#pragma unroll
    for (int m = 1; m < 16; m <<= 1) v = fmaxf(v, __shfl_xor(v, m, 32));
    return v;
}

// ---------------------------------------------------------------------------
// Conversion kernels
// ---------------------------------------------------------------------------
__global__ void cvt_f32_to_bf16(const float* __restrict__ in, bf16* __restrict__ out, int n) {
    int i = blockIdx.x * blockDim.x + threadIdx.x;
    if (i < n) out[i] = (bf16)in[i];
}

// W [K x Nw] f32 row-major  ->  Wt [Nw x K] bf16 row-major
__global__ void cvt_transpose_bf16(const float* __restrict__ in, bf16* __restrict__ out,
                                   int K, int Nw) {
    __shared__ float tile[32][33];
    int k0 = blockIdx.y * 32, n0 = blockIdx.x * 32;
    int tx = threadIdx.x, ty = threadIdx.y;   // 32 x 8
#pragma unroll
    for (int j = 0; j < 32; j += 8)
        tile[ty + j][tx] = in[(size_t)(k0 + ty + j) * Nw + n0 + tx];
    __syncthreads();
#pragma unroll
    for (int j = 0; j < 32; j += 8)
        out[(size_t)(n0 + ty + j) * K + k0 + tx] = (bf16)tile[tx][ty + j];
}

// ---------------------------------------------------------------------------
// bf16 WMMA GEMM: C[8192 x 1024] = A[8192 x 1024] @ Bt^T   (Bt[n][k] = W[k][n]).
// Each wave: 32x64 tile (2 M-frags x 4 N-frags = 8 WMMA per K-step of 32).
// MODE 0: bf16 row-major out, *scale
// MODE 1: bf16 transposed out per batch: out[(b*1024+col)*4096 + m]   (for V^T)
// MODE 2: f32 row-major out + bias
// ---------------------------------------------------------------------------
template <int MODE>
__global__ void __launch_bounds__(256)
gemm_bf16(const bf16* __restrict__ A, const bf16* __restrict__ Bt,
          void* __restrict__ Cout, const float* __restrict__ bias, float scale) {
    const int lane  = threadIdx.x & 31;
    const int wave  = threadIdx.x >> 5;
    const int col15 = lane & 15;
    const int half  = lane >> 4;
    const int row0  = blockIdx.x * 32;                 // 256 m-tiles
    const int col0  = (blockIdx.y * 8 + wave) * 64;    // 16 n-tiles of 64

    v8f acc[2][4];
#pragma unroll
    for (int i = 0; i < 2; ++i)
#pragma unroll
        for (int j = 0; j < 4; ++j)
#pragma unroll
            for (int r = 0; r < 8; ++r) acc[i][j][r] = 0.0f;

    for (int k0 = 0; k0 < DMODEL; k0 += 32) {
        v16bf aF[2], bF[4];
#pragma unroll
        for (int i = 0; i < 2; ++i)
            aF[i] = ldfragA(A + (size_t)(row0 + i * 16 + col15) * DMODEL + k0, half);
#pragma unroll
        for (int j = 0; j < 4; ++j)
            bF[j] = ldfrag16(Bt + (size_t)(col0 + j * 16 + col15) * DMODEL + k0 + 16 * half);
#pragma unroll
        for (int i = 0; i < 2; ++i)
#pragma unroll
            for (int j = 0; j < 4; ++j)
                acc[i][j] = wmma_bf16(aF[i], bF[j], acc[i][j]);
    }

#pragma unroll
    for (int i = 0; i < 2; ++i)
#pragma unroll
        for (int j = 0; j < 4; ++j) {
            int colg = col0 + j * 16 + col15;
#pragma unroll
            for (int r = 0; r < 8; ++r) {
                int rowg = row0 + i * 16 + (half ? 8 + r : r);
                float v = acc[i][j][r] * scale;
                if (MODE == 0) {
                    ((bf16*)Cout)[(size_t)rowg * DMODEL + colg] = (bf16)v;
                } else if (MODE == 1) {
                    int b = rowg >> 12;          // rowg / 4096
                    int m = rowg & 4095;
                    ((bf16*)Cout)[((size_t)(b * DMODEL + colg)) * SEQ_M + m] = (bf16)v;
                } else {
                    ((float*)Cout)[(size_t)rowg * DMODEL + colg] = v + bias[colg];
                }
            }
        }
}

// ---------------------------------------------------------------------------
// Flash attention: one wave per (b, h, 32-query-row tile); streams 32 keys/iter.
// Two 16-row q-tiles share every K/V fragment -> 18 WMMA per 32-key chunk
// (8 for S, 2 for P row-sums via ones-matrix, 8 for P@V).
// __launch_bounds__(256,1): relax occupancy target so nothing spills.
// ---------------------------------------------------------------------------
__global__ void __launch_bounds__(256, 1)
flash_attn(const bf16* __restrict__ q, const bf16* __restrict__ k,
           const bf16* __restrict__ vt, bf16* __restrict__ o) {
    __shared__ bf16 pshared[8][2][16 * 32];   // per-wave, per-qtile P (16x32)
    const int lane  = threadIdx.x & 31;
    const int wave  = threadIdx.x >> 5;
    const int col15 = lane & 15;
    const int half  = lane >> 4;

    int tile = blockIdx.x * 8 + wave;          // 4096 tiles total
    const int nt = tile & 127;                 // N/32 = 128
    const int h  = (tile >> 7) & 15;
    const int b  = tile >> 11;

    // ones B-fragment: rowsum(P) = P @ ones(32x16), same C/D layout as O
    v16bf onesF;
#pragma unroll
    for (int i = 0; i < 16; ++i) onesF[i] = (bf16)1.0f;

    // Q A-fragments: 2 q-tiles x (dims 0-31, 32-63)
    v16bf qA[2][2];
#pragma unroll
    for (int qt = 0; qt < 2; ++qt) {
        const bf16* qrow =
            q + (size_t)(b * SEQ_N + nt * 32 + qt * 16 + col15) * DMODEL + h * DHEAD;
        qA[qt][0] = ldfragA(qrow, half);
        qA[qt][1] = ldfragA(qrow + 32, half);
    }

    v8f O[2][4];
    float m_[2][8], l_[2][8];
#pragma unroll
    for (int qt = 0; qt < 2; ++qt) {
#pragma unroll
        for (int t = 0; t < 4; ++t)
#pragma unroll
            for (int r = 0; r < 8; ++r) O[qt][t][r] = 0.0f;
#pragma unroll
        for (int r = 0; r < 8; ++r) { m_[qt][r] = -1e30f; l_[qt][r] = 0.0f; }
    }

    const bf16* kbase = k + (size_t)(b * SEQ_M) * DMODEL + h * DHEAD;
    const bf16* vbase = vt + (size_t)(b * DMODEL + h * DHEAD) * SEQ_M;

    for (int m0 = 0; m0 < SEQ_M; m0 += 32) {
        // S tiles: 2 q-tiles x 2 key-groups; K fragments shared across q-tiles
        v8f S[2][2];
#pragma unroll
        for (int g = 0; g < 2; ++g) {
            const bf16* kr = kbase + (size_t)(m0 + g * 16 + col15) * DMODEL;
            v16bf b0 = ldfrag16(kr + 16 * half);        // dims 0..31
            v16bf b1 = ldfrag16(kr + 32 + 16 * half);   // dims 32..63
#pragma unroll
            for (int qt = 0; qt < 2; ++qt) {
                v8f s;
#pragma unroll
                for (int r = 0; r < 8; ++r) s[r] = 0.0f;
                s = wmma_bf16(qA[qt][0], b0, s);
                s = wmma_bf16(qA[qt][1], b1, s);
                S[qt][g] = s;
            }
        }

        // Online softmax (max via lane shuffles; sums via ones-WMMA below)
        float corr_[2][8];
#pragma unroll
        for (int qt = 0; qt < 2; ++qt) {
            bf16* pl = &pshared[wave][qt][0];
#pragma unroll
            for (int r = 0; r < 8; ++r) {
                float mx   = hmax16(fmaxf(S[qt][0][r], S[qt][1][r]));
                float mnew = fmaxf(m_[qt][r], mx);
                float corr = __expf(m_[qt][r] - mnew);
                float p0   = __expf(S[qt][0][r] - mnew);
                float p1   = __expf(S[qt][1][r] - mnew);
                corr_[qt][r] = corr;
                m_[qt][r]    = mnew;
                O[qt][0][r] *= corr; O[qt][1][r] *= corr;
                O[qt][2][r] *= corr; O[qt][3][r] *= corr;
                int row = half ? 8 + r : r;
                pl[row * 32 + col15]      = (bf16)p0;
                pl[row * 32 + 16 + col15] = (bf16)p1;
            }
        }
        // per-wave LDS RAW: DS ops are in-order within a wave; be explicit anyway
        asm volatile("s_wait_dscnt 0x0" ::: "memory");

        // P as A-fragments (K = local key index)
        v16bf pA[2];
#pragma unroll
        for (int qt = 0; qt < 2; ++qt)
            pA[qt] = ldfragA(&pshared[wave][qt][0] + col15 * 32, half);

        // Row-sums of (bf16-rounded) P via one WMMA per q-tile
#pragma unroll
        for (int qt = 0; qt < 2; ++qt) {
            v8f z;
#pragma unroll
            for (int r = 0; r < 8; ++r) z[r] = 0.0f;
            v8f ls = wmma_bf16(pA[qt], onesF, z);
#pragma unroll
            for (int r = 0; r < 8; ++r)
                l_[qt][r] = l_[qt][r] * corr_[qt][r] + ls[r];
        }

        // O += P @ V  (4 N-tiles of 16 dims; V fragments shared across q-tiles)
#pragma unroll
        for (int t = 0; t < 4; ++t) {
            const bf16* vr = vbase + (size_t)(t * 16 + col15) * SEQ_M + m0 + 16 * half;
            v16bf vB = ldfrag16(vr);
            O[0][t] = wmma_bf16(pA[0], vB, O[0][t]);
            O[1][t] = wmma_bf16(pA[1], vB, O[1][t]);
        }
    }

    // normalize + store
#pragma unroll
    for (int qt = 0; qt < 2; ++qt)
#pragma unroll
        for (int t = 0; t < 4; ++t) {
            int col = h * DHEAD + t * 16 + col15;
#pragma unroll
            for (int r = 0; r < 8; ++r) {
                int row = nt * 32 + qt * 16 + (half ? 8 + r : r);
                o[(size_t)(b * SEQ_N + row) * DMODEL + col] =
                    (bf16)(O[qt][t][r] / l_[qt][r]);
            }
        }
}

// ---------------------------------------------------------------------------
// Launch
// ---------------------------------------------------------------------------
extern "C" void kernel_launch(void* const* d_in, const int* in_sizes, int n_in,
                              void* d_out, int out_size, void* d_ws, size_t ws_size,
                              hipStream_t stream) {
    const float* x   = (const float*)d_in[0];
    const float* ctx = (const float*)d_in[1];
    const float* Wq  = (const float*)d_in[2];
    const float* Wk  = (const float*)d_in[3];
    const float* Wv  = (const float*)d_in[4];
    const float* Wo  = (const float*)d_in[5];
    const float* bo  = (const float*)d_in[6];
    float* out = (float*)d_out;

    char* ws = (char*)d_ws;
    size_t off = 0;
    auto carve = [&](size_t bytes) { void* p = ws + off; off += (bytes + 255) & ~size_t(255); return p; };

    bf16* xb  = (bf16*)carve((size_t)ROWS * DMODEL * 2);
    bf16* cb  = (bf16*)carve((size_t)ROWS * DMODEL * 2);
    bf16* wqT = (bf16*)carve((size_t)DMODEL * DMODEL * 2);
    bf16* wkT = (bf16*)carve((size_t)DMODEL * DMODEL * 2);
    bf16* wvT = (bf16*)carve((size_t)DMODEL * DMODEL * 2);
    bf16* woT = (bf16*)carve((size_t)DMODEL * DMODEL * 2);
    bf16* qb  = (bf16*)carve((size_t)ROWS * DMODEL * 2);
    bf16* kb  = (bf16*)carve((size_t)ROWS * DMODEL * 2);
    bf16* vtb = (bf16*)carve((size_t)BATCH * DMODEL * SEQ_M * 2);
    bf16* ab  = (bf16*)carve((size_t)ROWS * DMODEL * 2);
    (void)ws_size; (void)in_sizes; (void)n_in; (void)out_size;

    const int nElem = ROWS * DMODEL;   // 8 388 608
    cvt_f32_to_bf16<<<nElem / 256, 256, 0, stream>>>(x, xb, nElem);
    cvt_f32_to_bf16<<<nElem / 256, 256, 0, stream>>>(ctx, cb, nElem);

    dim3 tb(32, 8), tg(DMODEL / 32, DMODEL / 32);
    cvt_transpose_bf16<<<tg, tb, 0, stream>>>(Wq, wqT, DMODEL, DMODEL);
    cvt_transpose_bf16<<<tg, tb, 0, stream>>>(Wk, wkT, DMODEL, DMODEL);
    cvt_transpose_bf16<<<tg, tb, 0, stream>>>(Wv, wvT, DMODEL, DMODEL);
    cvt_transpose_bf16<<<tg, tb, 0, stream>>>(Wo, woT, DMODEL, DMODEL);

    dim3 gg(ROWS / 32, (DMODEL / 64) / 8);   // (256, 2) blocks, 8 waves each
    const float scale = 0.125f;              // DIM_HEAD^-0.5
    gemm_bf16<0><<<gg, 256, 0, stream>>>(xb, wqT, qb,  nullptr, scale);  // Q (pre-scaled)
    gemm_bf16<0><<<gg, 256, 0, stream>>>(cb, wkT, kb,  nullptr, 1.0f);   // K
    gemm_bf16<1><<<gg, 256, 0, stream>>>(cb, wvT, vtb, nullptr, 1.0f);   // V (transposed)

    flash_attn<<<(BATCH * HEADS * (SEQ_N / 32)) / 8, 256, 0, stream>>>(qb, kb, vtb, ab);

    gemm_bf16<2><<<gg, 256, 0, stream>>>(ab, woT, out, bo, 1.0f);        // out proj + bias
}
